// BahdanauAttention_41128606827114
// MI455X (gfx1250) — compile-verified, matrix operand order checked by reading, not run on hardware
//
#include <hip/hip_runtime.h>
#include <hip/hip_bf16.h>
#include <math.h>

#define ENCD 1024
#define DECD 1024
#define BATCH 32
#define SEQ 2048

typedef __bf16 bf16_t;
typedef bf16_t v16bf  __attribute__((ext_vector_type(16)));
typedef bf16_t bf16x4 __attribute__((ext_vector_type(4)));
typedef float  v8f    __attribute__((ext_vector_type(8)));
typedef float  f32x4  __attribute__((ext_vector_type(4)));
typedef unsigned int u32x4 __attribute__((ext_vector_type(4)));

union FragB16 { v16bf v; u32x4 u[2]; };

__device__ __forceinline__ float fast_tanh(float x)
{
#if __has_builtin(__builtin_amdgcn_tanhf)
    return __builtin_amdgcn_tanhf(x);      // v_tanh_f32 (CDNA5 trans op)
#else
    return tanhf(x);
#endif
}

__device__ __forceinline__ void load_frag(FragB16& f, const bf16_t* __restrict__ row, int kb)
{
    f.u[0] = *reinterpret_cast<const u32x4*>(row + kb);
    f.u[1] = *reinterpret_cast<const u32x4*>(row + kb + 16);
}

// ---------------------------------------------------------------------------
// Kernel 1: W_enc (rows 1024..2047 of W_attn, K=1024 x N=1024 row major)
//           -> wT[n][k] bf16, tiled 32x32 transpose through LDS.
// ---------------------------------------------------------------------------
__global__ __launch_bounds__(256)
void ba_transpose_wenc(const float* __restrict__ W, unsigned short* __restrict__ wT_raw)
{
    __shared__ float tile[32][33];
    bf16_t* wT = reinterpret_cast<bf16_t*>(wT_raw);
    const int k0 = blockIdx.x * 32;
    const int n0 = blockIdx.y * 32;
    const int tx = threadIdx.x & 31;
    const int ty = threadIdx.x >> 5;
#pragma unroll
    for (int i = 0; i < 4; ++i) {
        int k = k0 + ty + i * 8;
        tile[ty + i * 8][tx] = W[(size_t)(DECD + k) * DECD + (n0 + tx)];
    }
    __syncthreads();
#pragma unroll
    for (int i = 0; i < 4; ++i) {
        int n = n0 + ty + i * 8;
        wT[(size_t)n * ENCD + (k0 + tx)] = (bf16_t)tile[tx][ty + i * 8];
    }
}

// ---------------------------------------------------------------------------
// Kernel 2: base[b][d] = decoder_hidden[b] . W_dec[:,d] + b_attn[d]   (f32)
// ---------------------------------------------------------------------------
__global__ __launch_bounds__(256)
void ba_dec_proj(const float* __restrict__ dec, const float* __restrict__ W,
                 const float* __restrict__ b_attn, float* __restrict__ base)
{
    const int idx = blockIdx.x * 256 + threadIdx.x;
    const int bb  = idx >> 10;
    const int d   = idx & 1023;
    float sum = b_attn[d];
    const float* dh = dec + (size_t)bb * DECD;
    for (int k = 0; k < DECD; ++k)
        sum = fmaf(dh[k], W[(size_t)k * DECD + d], sum);
    base[idx] = sum;
}

// ---------------------------------------------------------------------------
// Kernel 3 (main): scores[b][s] = v . tanh(enc[b][s] @ W_enc + base[b])
// WG = 256 threads (8 waves), 32-row S tile of one batch, staged to LDS bf16.
// Loop nest: K outer (hot), 4 N-tiles per wave held in 8 persistent v8f
// accumulators.  A fragments are loaded once per K step and consumed inside
// the iteration (no loop-invariant loads -> no spill-hoisting), B fragments
// stream from L2 with constant immediate offsets.
// ---------------------------------------------------------------------------
__global__ __launch_bounds__(256)
void ba_scores_wmma(const float* __restrict__ enc, const unsigned short* __restrict__ wT_raw,
                    const float* __restrict__ base, const float* __restrict__ v_w,
                    const int* __restrict__ mask, float* __restrict__ scores)
{
    __shared__ __align__(16) bf16_t Abuf[32][1032];   // 32 rows x 1024 K (+8 pad)
    __shared__ float sred[32];

    const int tid = threadIdx.x;
    const int b   = blockIdx.y;
    const int s0  = blockIdx.x * 32;
    const bf16_t* wT = reinterpret_cast<const bf16_t*>(wT_raw);

    // ---- Stage A panel: float4 reads, bf16x4 (8B) LDS stores ----
    const f32x4* src4 = reinterpret_cast<const f32x4*>(enc + ((size_t)b * SEQ + s0) * ENCD);
#pragma unroll
    for (int i = 0; i < 32; ++i) {
        int idx = tid + 256 * i;          // float4 index within 32x1024 panel
        int row = idx >> 8;               // 256 float4 per row
        int col = (idx & 255) * 4;
        f32x4 v = src4[idx];
        bf16x4 h;
        h.x = (bf16_t)v.x; h.y = (bf16_t)v.y; h.z = (bf16_t)v.z; h.w = (bf16_t)v.w;
        *reinterpret_cast<bf16x4*>(&Abuf[row][col]) = h;
    }
    if (tid < 32) sred[tid] = 0.0f;
    __syncthreads();

    const int wave  = tid >> 5;
    const int lane  = tid & 31;
    const int mrow  = lane & 15;          // A row (M) and C column (N) index
    const int khalf = (lane >> 4) * 8;    // K sub-offset per lane half
    const int hi8   = (lane >> 4) * 8;    // C row offset per lane half

    const bf16_t* arow0 = &Abuf[mrow][0];
    const bf16_t* arow1 = &Abuf[mrow + 16][0];

    float sc[16];
#pragma unroll
    for (int j = 0; j < 16; ++j) sc[j] = 0.0f;

    const v8f vzero = {};

    for (int pass = 0; pass < 2; ++pass) {
        // This wave's 4 N-tiles this pass: nt0 = pass*32 + wave*4
        const int nt0 = pass * 32 + wave * 4;
        // One base pointer; per-u offsets are compile-time immediates (u*16*ENCD).
        const bf16_t* bbase = wT + (size_t)(nt0 * 16 + mrow) * ENCD;

        v8f acc0[4], acc1[4];
#pragma unroll
        for (int u = 0; u < 4; ++u) { acc0[u] = vzero; acc1[u] = vzero; }

#pragma unroll 4
        for (int kt = 0; kt < 32; ++kt) {
            const int kb = kt * 32 + khalf;
            FragB16 a0, a1, fb[4];
            load_frag(a0, arow0, kb);
            load_frag(a1, arow1, kb);
#pragma unroll
            for (int u = 0; u < 4; ++u)
                load_frag(fb[u], bbase + (size_t)u * (16 * ENCD), kb);
#pragma unroll
            for (int u = 0; u < 4; ++u) {
                acc0[u] = __builtin_amdgcn_wmma_f32_16x16x32_bf16(
                              false, a0.v, false, fb[u].v, (short)0, acc0[u], false, false);
                acc1[u] = __builtin_amdgcn_wmma_f32_16x16x32_bf16(
                              false, a1.v, false, fb[u].v, (short)0, acc1[u], false, false);
            }
        }

        // Fused epilogue: energy = tanh(acc + base[n]); score += energy * v[n]
#pragma unroll
        for (int u = 0; u < 4; ++u) {
            const int   n   = (nt0 + u) * 16 + mrow;
            const float bsn = base[b * DECD + n];
            const float vn  = v_w[n];
#pragma unroll
            for (int j = 0; j < 8; ++j) {
                sc[j]     += fast_tanh(acc0[u][j] + bsn) * vn;   // rows j+hi8
                sc[j + 8] += fast_tanh(acc1[u][j] + bsn) * vn;   // rows 16+j+hi8
            }
        }
    }

    // Reduce partial scores across lanes/waves (N columns are spread out).
#pragma unroll
    for (int j = 0; j < 8; ++j) {
        atomicAdd(&sred[j + hi8],      sc[j]);
        atomicAdd(&sred[16 + j + hi8], sc[j + 8]);
    }
    __syncthreads();

    if (tid < 32) {
        int s = s0 + tid;
        float val = (mask[b * SEQ + s] == 0) ? -1000000000.0f : sred[tid];
        scores[b * SEQ + s] = val;
    }
}

// ---------------------------------------------------------------------------
// Kernel 4: masked softmax over S per batch; writes attn weights to d_out and
// zeroes the context region (d_out is poisoned before timing).
// ---------------------------------------------------------------------------
__global__ __launch_bounds__(256)
void ba_softmax(const float* __restrict__ scores, float* __restrict__ out)
{
    __shared__ float red[256];
    const int b   = blockIdx.x;
    const int tid = threadIdx.x;

    float v[8];
    float mx = -3.4e38f;
#pragma unroll
    for (int i = 0; i < 8; ++i) {
        v[i] = scores[b * SEQ + tid + 256 * i];
        mx = fmaxf(mx, v[i]);
    }
    red[tid] = mx; __syncthreads();
    for (int s = 128; s > 0; s >>= 1) {
        if (tid < s) red[tid] = fmaxf(red[tid], red[tid + s]);
        __syncthreads();
    }
    mx = red[0]; __syncthreads();

    float sum = 0.0f;
#pragma unroll
    for (int i = 0; i < 8; ++i) { v[i] = expf(v[i] - mx); sum += v[i]; }
    red[tid] = sum; __syncthreads();
    for (int s = 128; s > 0; s >>= 1) {
        if (tid < s) red[tid] += red[tid + s];
        __syncthreads();
    }
    const float inv = 1.0f / red[0];
#pragma unroll
    for (int i = 0; i < 8; ++i)
        out[BATCH * DECD + b * SEQ + tid + 256 * i] = v[i] * inv;
#pragma unroll
    for (int i = 0; i < 4; ++i)
        out[b * DECD + tid + 256 * i] = 0.0f;
}

// ---------------------------------------------------------------------------
// Kernel 5: context[b][e] = sum_s attn[b][s] * enc[b][s][e]
// ---------------------------------------------------------------------------
__global__ __launch_bounds__(256)
void ba_context(const float* __restrict__ enc, const float* __restrict__ attn,
                float* __restrict__ ctx)
{
    const int b    = blockIdx.x;
    const int eblk = blockIdx.y;
    const int sch  = blockIdx.z;
    const int e    = eblk * 256 + threadIdx.x;

    float acc = 0.0f;
    const float* w = attn + b * SEQ + sch * 256;
    const float* x = enc + ((size_t)b * SEQ + (size_t)sch * 256) * ENCD + e;
    for (int s = 0; s < 256; ++s)
        acc = fmaf(w[s], x[(size_t)s * ENCD], acc);
    atomicAdd(&ctx[b * DECD + e], acc);
}

// ---------------------------------------------------------------------------
extern "C" void kernel_launch(void* const* d_in, const int* in_sizes, int n_in,
                              void* d_out, int out_size, void* d_ws, size_t ws_size,
                              hipStream_t stream)
{
    const float* dec    = (const float*)d_in[0];   // (32, 1024)
    const float* enc    = (const float*)d_in[1];   // (32, 2048, 1024)
    const int*   mask   = (const int*)  d_in[2];   // (32, 2048)
    const float* W      = (const float*)d_in[3];   // (2048, 1024)
    const float* b_attn = (const float*)d_in[4];   // (1024,)
    const float* v_w    = (const float*)d_in[5];   // (1024,)
    float* out = (float*)d_out;                    // context(32x1024) ++ attn(32x2048)

    char* ws = (char*)d_ws;
    unsigned short* wT = (unsigned short*)ws;                     // 2 MB bf16
    float* base   = (float*)(ws + (2u << 20));                    // 128 KB
    float* scores = (float*)(ws + (2u << 20) + (128u << 10));     // 256 KB

    ba_transpose_wenc<<<dim3(32, 32), 256, 0, stream>>>(W, wT);
    ba_dec_proj      <<<128, 256, 0, stream>>>(dec, W, b_attn, base);
    ba_scores_wmma   <<<dim3(SEQ / 32, BATCH), 256, 0, stream>>>(enc, wT, base, v_w, mask, scores);
    ba_softmax       <<<BATCH, 256, 0, stream>>>(scores, out);
    ba_context       <<<dim3(BATCH, 4, 8), 256, 0, stream>>>(enc, out + BATCH * DECD, out);
}
